// CrossLayerTranscoder_51453708206742
// MI455X (gfx1250) — matrix operand order repaired and checked by reading, not compiled, not run on hardware
//
#include <hip/hip_runtime.h>
#include <hip/hip_bf16.h>
#include <stdint.h>

// -----------------------------------------------------------------------------
// CrossLayerTranscoder on MI455X (gfx1250), wave32 + v_wmma_f32_16x16x32_bf16.
//
//   encoder : feats[l]  = JumpReLU(res[l] @ We[l]^T + be[l])        (8 GEMMs)
//   decoder : recon[j]  = sum_{l<=j} feats[l] @ Wd[tri(l,j)]^T + bias (1 fat-K
//             GEMM per output layer j, K = (j+1)*4096)
//
// fp32 operands are converted to bf16 with the native packed cvt while staging
// into LDS; WMMA accumulates in fp32. 128x128 block tile, 8 waves of 32x64,
// K-step 32, double-buffered LDS with one barrier per K-step.
// -----------------------------------------------------------------------------

typedef __bf16 v16bf __attribute__((ext_vector_type(16)));
typedef __bf16 bf2   __attribute__((ext_vector_type(2)));
typedef float  v8f   __attribute__((ext_vector_type(8)));

#define L_LAYERS 8
#define H_DIM    1024
#define F_DIM    4096
#define BS       2048                               // B*S tokens
#define RECON_ELEMS (L_LAYERS * BS * H_DIM)         // 16,777,216 floats

#define TM 128
#define TN 128
#define TK 32
#define PITCH_B 80                                  // 32 bf16 (64B) + 16B pad
#define TILE_BYTES (128 * PITCH_B)                  // 10240 B per operand tile
#define BUF_BYTES  (2 * TILE_BYTES)                 // A + B

// fp32 -> packed 2x bf16 using the native v_cvt_pk_bf16_f32 when available
__device__ __forceinline__ uint32_t pk2_bf16(float lo, float hi) {
#if __has_builtin(__builtin_amdgcn_cvt_pk_bf16_f32)
    bf2 t = __builtin_amdgcn_cvt_pk_bf16_f32(lo, hi);
    return __builtin_bit_cast(uint32_t, t);
#else
    bf2 t;
    t.x = (__bf16)lo;                               // fptrunc -> hw cvt
    t.y = (__bf16)hi;
    return __builtin_bit_cast(uint32_t, t);
#endif
}

struct Stage { uint32_t d[8]; };                    // 16 floats -> 16 bf16

__device__ __forceinline__ Stage ldconv(const float* __restrict__ p) {
    const float4* q = (const float4*)p;
    float4 f0 = q[0], f1 = q[1], f2 = q[2], f3 = q[3];
    Stage s;
    s.d[0] = pk2_bf16(f0.x, f0.y);  s.d[1] = pk2_bf16(f0.z, f0.w);
    s.d[2] = pk2_bf16(f1.x, f1.y);  s.d[3] = pk2_bf16(f1.z, f1.w);
    s.d[4] = pk2_bf16(f2.x, f2.y);  s.d[5] = pk2_bf16(f2.z, f2.w);
    s.d[6] = pk2_bf16(f3.x, f3.y);  s.d[7] = pk2_bf16(f3.z, f3.w);
    return s;
}

__device__ __forceinline__ void stg_tile(char* lds, int row, int half, const Stage& s) {
    uint4* dst = (uint4*)(lds + row * PITCH_B + half * 32);
    dst[0] = make_uint4(s.d[0], s.d[1], s.d[2], s.d[3]);
    dst[1] = make_uint4(s.d[4], s.d[5], s.d[6], s.d[7]);
}

// A fragment (16x32 bf16): lanes 0-15 -> rows, K chunks {0..7,16..23};
// lanes 16-31 -> same rows, K chunks {8..15,24..31}.
__device__ __forceinline__ v16bf load_a_frag(const char* ldsA, int rowBase, int lane) {
    const int lo = lane & 15, hi = lane >> 4;
    const char* p = ldsA + (rowBase + lo) * PITCH_B + hi * 16;
    union { v16bf v; uint4 q[2]; } u;
    u.q[0] = *(const uint4*)p;
    u.q[1] = *(const uint4*)(p + 32);
    return u.v;
}

// B fragment (32x16 bf16): lanes 0-15 -> cols, K chunks {0..7,16..23}; lanes 16-31 -> K chunks {8..15,24..31}.
__device__ __forceinline__ v16bf load_b_frag(const char* ldsB, int colBase, int lane) {
    const int lo = lane & 15, hi = lane >> 4;
    const char* p = ldsB + (colBase + lo) * PITCH_B + hi * 16;
    union { v16bf v; uint4 q[2]; } u;
    u.q[0] = *(const uint4*)p;
    u.q[1] = *(const uint4*)(p + 32);
    return u.v;
}

#define WMMA_BF16(A, B, C) \
    __builtin_amdgcn_wmma_f32_16x16x32_bf16(false, (A), false, (B), (short)0, (C), false, false)

// ---------------------------------------------------------------- encoder ---
__global__ __launch_bounds__(256)
void clt_encoder(const float* __restrict__ res, const float* __restrict__ We,
                 const float* __restrict__ be,  const float* __restrict__ thresh,
                 float* __restrict__ out) {
    __shared__ __align__(16) char smem[2 * BUF_BYTES];

    const int l  = blockIdx.z;
    const int m0 = blockIdx.y * TM;
    const int n0 = blockIdx.x * TN;

    const int tid  = threadIdx.x;
    const int lrow = tid >> 1;                       // 0..127
    const int half = tid & 1;

    const float* Abase = res + ((size_t)l * BS + (m0 + lrow)) * H_DIM + half * 16;
    const float* Bbase = We  + ((size_t)l * F_DIM + (n0 + lrow)) * H_DIM + half * 16;

    const int wave = tid >> 5;
    const int lane = tid & 31;
    const int mw   = (wave & 3) * 32;
    const int nw   = (wave >> 2) * 64;

    const v8f vzero = {0.f, 0.f, 0.f, 0.f, 0.f, 0.f, 0.f, 0.f};
    v8f acc[2][4];
#pragma unroll
    for (int mi = 0; mi < 2; ++mi)
#pragma unroll
        for (int nf = 0; nf < 4; ++nf) acc[mi][nf] = vzero;

    const int nK = H_DIM / TK;                        // 32 K-steps

    Stage ga = ldconv(Abase);
    Stage gb = ldconv(Bbase);

    for (int kt = 0; kt < nK; ++kt) {
        char* buf = smem + (kt & 1) * BUF_BYTES;
        if (kt == 0) {
            stg_tile(buf, lrow, half, ga);
            stg_tile(buf + TILE_BYTES, lrow, half, gb);
        }
        __syncthreads();

        if (kt + 1 < nK) {
            ga = ldconv(Abase + (size_t)(kt + 1) * TK);
            gb = ldconv(Bbase + (size_t)(kt + 1) * TK);
            __builtin_prefetch(Abase + (size_t)(kt + 3) * TK, 0, 1);
            __builtin_prefetch(Bbase + (size_t)(kt + 3) * TK, 0, 1);
        }

        const char* bufA = buf;
        const char* bufB = buf + TILE_BYTES;
        v16bf a[2], b[4];
        a[0] = load_a_frag(bufA, mw,      lane);
        a[1] = load_a_frag(bufA, mw + 16, lane);
#pragma unroll
        for (int nf = 0; nf < 4; ++nf) b[nf] = load_b_frag(bufB, nw + nf * 16, lane);
#pragma unroll
        for (int mi = 0; mi < 2; ++mi)
#pragma unroll
            for (int nf = 0; nf < 4; ++nf)
                acc[mi][nf] = WMMA_BF16(a[mi], b[nf], acc[mi][nf]);

        if (kt + 1 < nK) {
            char* nbuf = smem + ((kt + 1) & 1) * BUF_BYTES;
            stg_tile(nbuf, lrow, half, ga);
            stg_tile(nbuf + TILE_BYTES, lrow, half, gb);
        }
    }

    // epilogue: bias + JumpReLU, write feats region of d_out
    const float th = thresh[l];
    float* feats = out + RECON_ELEMS + (size_t)l * BS * F_DIM;
    const int lo = lane & 15, hi = lane >> 4;
#pragma unroll
    for (int nf = 0; nf < 4; ++nf) {
        const int n = n0 + nw + nf * 16 + lo;
        const float bias = be[(size_t)l * F_DIM + n];
#pragma unroll
        for (int mi = 0; mi < 2; ++mi) {
#pragma unroll
            for (int r = 0; r < 8; ++r) {
                const int m = m0 + mw + mi * 16 + hi * 8 + r;
                const float v = acc[mi][nf][r] + bias;
                feats[(size_t)m * F_DIM + n] = (v > th) ? v : 0.0f;
            }
        }
    }
}

// ---------------------------------------------------------------- decoder ---
__global__ __launch_bounds__(256)
void clt_decoder(const float* __restrict__ Wd, const float* __restrict__ bd,
                 float* __restrict__ out) {
    __shared__ __align__(16) char smem[2 * BUF_BYTES];

    const int j  = blockIdx.z;                        // output layer
    const int m0 = blockIdx.y * TM;
    const int n0 = blockIdx.x * TN;                   // over H

    const float* feats = out + RECON_ELEMS;

    const int tid  = threadIdx.x;
    const int lrow = tid >> 1;
    const int half = tid & 1;

    const int wave = tid >> 5;
    const int lane = tid & 31;
    const int mw   = (wave & 3) * 32;
    const int nw   = (wave >> 2) * 64;

    const v8f vzero = {0.f, 0.f, 0.f, 0.f, 0.f, 0.f, 0.f, 0.f};
    v8f acc[2][4];
#pragma unroll
    for (int mi = 0; mi < 2; ++mi)
#pragma unroll
        for (int nf = 0; nf < 4; ++nf) acc[mi][nf] = vzero;

    const int stepsPerL = F_DIM / TK;                 // 128
    const int nK = (j + 1) * stepsPerL;               // up to 1024

    auto aPtr = [&](int kk) -> const float* {
        const int l  = kk >> 7;
        const int kf = (kk & 127) * TK;
        return feats + ((size_t)l * BS + (m0 + lrow)) * F_DIM + kf + half * 16;
    };
    auto bPtr = [&](int kk) -> const float* {
        const int l  = kk >> 7;
        const int kf = (kk & 127) * TK;
        const int p  = l * L_LAYERS - (l * (l - 1)) / 2 + (j - l);   // tri index
        return Wd + ((size_t)p * H_DIM + (n0 + lrow)) * F_DIM + kf + half * 16;
    };

    Stage ga = ldconv(aPtr(0));
    Stage gb = ldconv(bPtr(0));

    for (int kt = 0; kt < nK; ++kt) {
        char* buf = smem + (kt & 1) * BUF_BYTES;
        if (kt == 0) {
            stg_tile(buf, lrow, half, ga);
            stg_tile(buf + TILE_BYTES, lrow, half, gb);
        }
        __syncthreads();

        if (kt + 1 < nK) {
            ga = ldconv(aPtr(kt + 1));
            gb = ldconv(bPtr(kt + 1));
            if (kt + 3 < nK) {
                __builtin_prefetch(aPtr(kt + 3), 0, 1);
                __builtin_prefetch(bPtr(kt + 3), 0, 1);
            }
        }

        const char* bufA = buf;
        const char* bufB = buf + TILE_BYTES;
        v16bf a[2], b[4];
        a[0] = load_a_frag(bufA, mw,      lane);
        a[1] = load_a_frag(bufA, mw + 16, lane);
#pragma unroll
        for (int nf = 0; nf < 4; ++nf) b[nf] = load_b_frag(bufB, nw + nf * 16, lane);
#pragma unroll
        for (int mi = 0; mi < 2; ++mi)
#pragma unroll
            for (int nf = 0; nf < 4; ++nf)
                acc[mi][nf] = WMMA_BF16(a[mi], b[nf], acc[mi][nf]);

        if (kt + 1 < nK) {
            char* nbuf = smem + ((kt + 1) & 1) * BUF_BYTES;
            stg_tile(nbuf, lrow, half, ga);
            stg_tile(nbuf + TILE_BYTES, lrow, half, gb);
        }
    }

    // epilogue: recon[j] = acc + sum_{l<=j} bd[tri(l,j)]
    const int lo = lane & 15, hi = lane >> 4;
#pragma unroll
    for (int nf = 0; nf < 4; ++nf) {
        const int n = n0 + nw + nf * 16 + lo;
        float bsum = 0.0f;
        for (int l = 0; l <= j; ++l) {
            const int p = l * L_LAYERS - (l * (l - 1)) / 2 + (j - l);
            bsum += bd[(size_t)p * H_DIM + n];
        }
#pragma unroll
        for (int mi = 0; mi < 2; ++mi) {
#pragma unroll
            for (int r = 0; r < 8; ++r) {
                const int m = m0 + mw + mi * 16 + hi * 8 + r;
                out[((size_t)j * BS + m) * H_DIM + n] = acc[mi][nf][r] + bsum;
            }
        }
    }
}

// ------------------------------------------------------------------ launch ---
extern "C" void kernel_launch(void* const* d_in, const int* in_sizes, int n_in,
                              void* d_out, int out_size, void* d_ws, size_t ws_size,
                              hipStream_t stream) {
    (void)in_sizes; (void)n_in; (void)out_size; (void)d_ws; (void)ws_size;
    const float* res    = (const float*)d_in[0];
    const float* We     = (const float*)d_in[1];
    const float* be     = (const float*)d_in[2];
    const float* Wd     = (const float*)d_in[3];
    const float* bd     = (const float*)d_in[4];
    const float* thresh = (const float*)d_in[5];
    float* out = (float*)d_out;

    dim3 ge(F_DIM / TN, BS / TM, L_LAYERS);           // 32 x 16 x 8
    clt_encoder<<<ge, dim3(256), 0, stream>>>(res, We, be, thresh, out);

    dim3 gd(H_DIM / TN, BS / TM, L_LAYERS);           // 8 x 16 x 8
    clt_decoder<<<gd, dim3(256), 0, stream>>>(Wd, bd, out);
}
